// WeightOnlyLinear_18975165514260
// MI455X (gfx1250) — matrix operand order repaired and checked by reading, not compiled, not run on hardware
//
#include <hip/hip_runtime.h>

typedef __attribute__((ext_vector_type(16))) _Float16 v16h;
typedef __attribute__((ext_vector_type(8)))  _Float16 h8;
typedef __attribute__((ext_vector_type(4)))  _Float16 h4;
typedef __attribute__((ext_vector_type(2)))  _Float16 h2;
typedef __attribute__((ext_vector_type(8)))  float    v8f;
typedef __attribute__((ext_vector_type(4)))  float    f4;

#define BLOCK_M   256
#define BLOCK_N   128
#define BLOCK_K   32
#define GROUPSIZE 128
#define LDS_K     40   // 32 + 8 halves pad -> 80B row stride (conflict-free, 16B aligned)

union Frag { v16h v; h8 h[2]; };
union H8x  { h8 v; h2 p[4]; };

__global__ __launch_bounds__(256)
void woq4_gemm_f16_wmma(const float* __restrict__ x,
                        const float* __restrict__ scales,
                        const float* __restrict__ bias,
                        const int*   __restrict__ qweight,
                        const int*   __restrict__ qzeros,
                        float*       __restrict__ out,
                        int tokens, int in_f, int out_f)
{
    __shared__ _Float16 lds_a[2][BLOCK_M * LDS_K];   // x tile (m, k), f16     (40 KB)
    __shared__ _Float16 lds_b[2][BLOCK_N * LDS_K];   // dequant W tile (n, k)  (20 KB)

    const int tid  = threadIdx.x;
    const int lane = tid & 31;
    const int wave = tid >> 5;

    const int bm = blockIdx.y * BLOCK_M;
    const int bn = blockIdx.x * BLOCK_N;

    // wave tile: 64(M) x 64(N); waves arranged 4(M) x 2(N)
    const int wm   = (wave >> 1) * 64;
    const int wn   = (wave & 1) * 64;
    const int half = lane >> 4;
    const int l15  = lane & 15;

    v8f acc[4][4];
    #pragma unroll
    for (int i = 0; i < 4; ++i)
        #pragma unroll
        for (int j = 0; j < 4; ++j)
            #pragma unroll
            for (int r = 0; r < 8; ++r)
                acc[i][j][r] = 0.0f;

    // ---- x loader: lane covers 16B segment xseg of row xrow+32u (fully coalesced) ----
    const int xrow = tid >> 3;                 // 0..31
    const int xseg = tid & 7;                  // 0..7 (4 floats each)
    const float* xbase = x + (size_t)(bm + xrow) * in_f + xseg * 4;

    // ---- weight loader: thread -> column n, packed words kwb, kwb+2 ----
    const int nw  = tid & 127;
    const int kwb = tid >> 7;                  // 0 or 1
    const int gn  = bn + nw;
    const int zsh = 4 * (gn & 7);
    const int qzc = gn >> 3;
    const int of8 = out_f >> 3;

    h4 xh[8]; unsigned qw[2]; h2 zoff = {}, scp = {};

    auto stage = [&](int k0) {                 // global -> registers (cvt early)
        #pragma unroll
        for (int u = 0; u < 8; ++u) {
            f4 v = *(const f4*)(xbase + (size_t)(32 * u) * in_f + k0);
            h4 t;
            #pragma unroll
            for (int e = 0; e < 4; ++e) t[e] = (_Float16)v[e];
            xh[u] = t;
        }
        const int g = k0 / GROUPSIZE;
        const float sc = scales[(size_t)g * out_f + gn];
        const unsigned qz = (unsigned)qzeros[(size_t)g * of8 + qzc];
        const int zp = (int)((qz >> zsh) & 15u) + 1;
        const _Float16 zh = (_Float16)(-(float)(1024 + zp));   // exact in f16
        const _Float16 sh = (_Float16)sc;
        zoff[0] = zh; zoff[1] = zh;
        scp[0]  = sh; scp[1]  = sh;
        const int kq0 = k0 >> 3;
        qw[0] = (unsigned)qweight[(size_t)(kq0 + kwb)     * out_f + gn];
        qw[1] = (unsigned)qweight[(size_t)(kq0 + kwb + 2) * out_f + gn];
    };

    auto commit = [&](int buf) {               // registers -> LDS
        #pragma unroll
        for (int u = 0; u < 8; ++u)
            *(h4*)(lds_a[buf] + (xrow + 32 * u) * LDS_K + xseg * 4) = xh[u];
        #pragma unroll
        for (int p = 0; p < 2; ++p) {
            const unsigned w = qw[p];
            const int kw = kwb + 2 * p;        // 0..3
            H8x d;
            #pragma unroll
            for (int pi = 0; pi < 4; ++pi) {
                // f16 magic: 0x6400|q == (f16)(1024+q) exactly; pk_add/pk_mul dequant
                unsigned u = w >> (8 * pi);
                unsigned bits = (u & 0xFu) | ((u & 0xF0u) << 12) | 0x64006400u;
                h2 val = __builtin_bit_cast(h2, bits);
                d.p[pi] = (val + zoff) * scp;
            }
            *(h8*)(lds_b[buf] + nw * LDS_K + kw * 8) = d.v;
        }
    };

    auto loadB = [&](int buf, int j) {
        const _Float16* rp = lds_b[buf] + (wn + j * 16 + l15) * LDS_K;
        // B 32x16: lanes 0-15 K{0..7,16..23}, lanes 16-31 K{8..15,24..31}
        Frag bf;
        bf.h[0] = *(const h8*)(rp + 8 * half);
        bf.h[1] = *(const h8*)(rp + 16 + 8 * half);
        return bf;
    };

    auto compute = [&](int buf) {              // 16 WMMA, 16 ds_load_b128
        Frag af[4];
        #pragma unroll
        for (int i = 0; i < 4; ++i) {
            const _Float16* rp = lds_a[buf] + (wm + i * 16 + l15) * LDS_K;
            // A 16-bit 16x32: lanes 0-15 K{0..7,16..23}, lanes 16-31 K{8..15,24..31}
            af[i].h[0] = *(const h8*)(rp + 8 * half);
            af[i].h[1] = *(const h8*)(rp + 16 + 8 * half);
        }
        Frag bnext = loadB(buf, 0);
        #pragma unroll
        for (int j = 0; j < 4; ++j) {
            Frag bcur = bnext;
            if (j < 3) bnext = loadB(buf, j + 1);   // pipeline B one column ahead
            #pragma unroll
            for (int i = 0; i < 4; ++i)
                acc[i][j] = __builtin_amdgcn_wmma_f32_16x16x32_f16(
                    /*neg_a=*/false, af[i].v,
                    /*neg_b=*/false, bcur.v,
                    /*c_mod=*/(short)0, acc[i][j],
                    /*reuse_a=*/false, /*reuse_b=*/false);
        }
    };

    const int ntiles = in_f / BLOCK_K;

    // software pipeline, last iteration peeled: no conditionals in hot loop
    stage(0);
    commit(0);
    __syncthreads();

    for (int t = 0; t < ntiles - 1; ++t) {
        const int cur = t & 1;
        stage((t + 1) * BLOCK_K);
        compute(cur);
        commit(cur ^ 1);
        __syncthreads();
    }
    compute((ntiles - 1) & 1);

    // ---------- epilogue: bias add + f32 store ----------
    float bv[4];
    #pragma unroll
    for (int j = 0; j < 4; ++j) bv[j] = bias[bn + wn + j * 16 + l15];
    #pragma unroll
    for (int i = 0; i < 4; ++i) {
        #pragma unroll
        for (int r = 0; r < 8; ++r) {
            const int m = bm + wm + i * 16 + 8 * half + r;
            float* op = out + (size_t)m * out_f + bn + wn + l15;
            #pragma unroll
            for (int j = 0; j < 4; ++j)
                op[j * 16] = acc[i][j][r] + bv[j];
        }
    }
}

extern "C" void kernel_launch(void* const* d_in, const int* in_sizes, int n_in,
                              void* d_out, int out_size, void* d_ws, size_t ws_size,
                              hipStream_t stream) {
    (void)n_in; (void)out_size; (void)d_ws; (void)ws_size;
    const float* x       = (const float*)d_in[0];
    const float* scales  = (const float*)d_in[1];
    const float* bias    = (const float*)d_in[2];
    const int*   qweight = (const int*)d_in[3];
    const int*   qzeros  = (const int*)d_in[4];
    float*       out     = (float*)d_out;

    const int out_f  = in_sizes[2];                                   // 12288
    const int in_f   = (int)(((long long)in_sizes[3] * 8) / out_f);   // 4096
    const int tokens = in_sizes[0] / in_f;                            // 8192

    dim3 grid(out_f / BLOCK_N, tokens / BLOCK_M);
    dim3 block(256);
    woq4_gemm_f16_wmma<<<grid, block, 0, stream>>>(
        x, scales, bias, qweight, qzeros, out, tokens, in_f, out_f);
}
